// GraphSAGEConv_74148315398476
// MI455X (gfx1250) — matrix operand order repaired and checked by reading, not compiled, not run on hardware
//
#include <hip/hip_runtime.h>
#include <hip/hip_bf16.h>

// ---------------------------------------------------------------------------
// GraphSAGE conv for MI455X (gfx1250, wave32, WMMA).
//   out = (scatter_mean(x_src[row] -> col)) @ W_l + b_l + x_tgt @ W_r
// Phase 1: fp32 atomic scatter-add (memory/atomic bound, L2-resident accum).
// Phase 2: bf16 hi/lo split WMMA GEMM (v_wmma_f32_16x16x32_bf16), fp32-class
//          accuracy via 3-term products, fused mean + bias + second GEMM.
// ---------------------------------------------------------------------------

typedef __attribute__((ext_vector_type(16))) __bf16 v16bf;
typedef __attribute__((ext_vector_type(8)))  float  v8f;

#define D_FEAT 256

union BF16x16 { v16bf v; unsigned short u[16]; };

static __device__ __forceinline__ unsigned short f2bf(float f) {
    unsigned int x = __float_as_uint(f);
    unsigned int r = x + 0x7FFFu + ((x >> 16) & 1u);   // round-to-nearest-even
    return (unsigned short)(r >> 16);
}
static __device__ __forceinline__ float bf2f(unsigned short b) {
    return __uint_as_float(((unsigned int)b) << 16);
}

// ---------------------------------------------------------------------------
// Zero the accumulator region (sums + counts) each launch (ws is poisoned).
// ---------------------------------------------------------------------------
__global__ void sage_zero_kernel(float* __restrict__ p, size_t n) {
    size_t i = (size_t)blockIdx.x * blockDim.x + threadIdx.x;
    size_t stride = (size_t)gridDim.x * blockDim.x;
    for (; i < n; i += stride) p[i] = 0.0f;
}

// ---------------------------------------------------------------------------
// Pre-swizzle a 256x256 fp32 weight (row-major [k][n]) into bf16 hi/lo arrays
// laid out as this file's WMMA B operand (32x16 K-major tiles, wave32, contiguous k order):
//   lanes 0-15 hold K=kt*32+e (e=0..15), lanes 16-31 hold K=kt*32+16+e,
//   column n = nt*16 + (lane&15). Lane's 16 bf16 values stored contiguously.
// Index: (((kt*16 + nt)*32 + lane)*16 + e)
// ---------------------------------------------------------------------------
__global__ void sage_prep_w_kernel(const float* __restrict__ W,
                                   unsigned short* __restrict__ whi,
                                   unsigned short* __restrict__ wlo) {
    int idx  = blockIdx.x * blockDim.x + threadIdx.x;   // 0..65535
    int e    = idx & 15;
    int lane = (idx >> 4) & 31;
    int nt   = (idx >> 9) & 15;
    int kt   = (idx >> 13) & 7;
    int K    = kt * 32 + ((lane >= 16) ? 16 : 0) + e;
    int n    = nt * 16 + (lane & 15);
    float v  = W[K * D_FEAT + n];
    unsigned short h = f2bf(v);
    whi[idx] = h;
    wlo[idx] = f2bf(v - bf2f(h));
}

// ---------------------------------------------------------------------------
// Edge scatter: one 256-thread block per edge; coalesced row load, coalesced
// fp32 atomic adds into sums[col], one count atomic per edge.
// ---------------------------------------------------------------------------
__global__ void __launch_bounds__(256)
sage_scatter_kernel(const float* __restrict__ x_src,
                    const int* __restrict__ edge_index,
                    float* __restrict__ sums, float* __restrict__ cnt,
                    int n_edges, int n_nodes) {
    int e = blockIdx.x;
    if (e >= n_edges) return;
    int r = edge_index[e];              // edge_index[0][e]  (source)
    int c = edge_index[n_edges + e];    // edge_index[1][e]  (target)
    if ((unsigned)r >= (unsigned)n_nodes || (unsigned)c >= (unsigned)n_nodes) return;
    const float* src = x_src + (size_t)r * D_FEAT;
    float*       dst = sums  + (size_t)c * D_FEAT;
    int t = threadIdx.x;
    float v = src[t];
    __hip_atomic_fetch_add(&dst[t], v, __ATOMIC_RELAXED, __HIP_MEMORY_SCOPE_AGENT);
    if (t == 0)
        __hip_atomic_fetch_add(&cnt[c], 1.0f, __ATOMIC_RELAXED, __HIP_MEMORY_SCOPE_AGENT);
}

// ---------------------------------------------------------------------------
// WMMA GEMM: one wave (32 threads) computes a 16-row x 64-col output tile.
//   grid.x = n_nodes/16 m-tiles, grid.y = 4 groups of 64 columns.
// A operands (aggr = sums/max(cnt,1), and x_tgt) are split hi/lo bf16 on the
// fly; B operands come pre-swizzled. 6 WMMAs per (kstep, ntile):
//   acc += Ah*Blh + Ah*Bll + Al*Blh  (aggr @ W_l, fp32-class accuracy)
//        + Th*Brh + Th*Brl + Tl*Brh  (x_tgt @ W_r)
// ---------------------------------------------------------------------------
__global__ void __launch_bounds__(32)
sage_gemm_kernel(const float* __restrict__ sums, const float* __restrict__ cnt,
                 const float* __restrict__ x_tgt, const float* __restrict__ b_l,
                 const unsigned short* __restrict__ wl_hi,
                 const unsigned short* __restrict__ wl_lo,
                 const unsigned short* __restrict__ wr_hi,
                 const unsigned short* __restrict__ wr_lo,
                 float* __restrict__ out, int n_nodes) {
    const int lane = threadIdx.x;                 // 0..31
    const int m0   = blockIdx.x * 16;
    const int ng   = blockIdx.y;                  // 64-column group
    const int n0   = ng * 64;

    // A layout used here (16-bit 16x32, contiguous k, not the ISA table order): lane L holds row M=L&15;
    // lanes<16: K = kk+{0..15}
    // lanes>=16: K = kk+{16..31}.
    const int mrow = lane & 15;
    int rowA = m0 + mrow;
    if (rowA >= n_nodes) rowA = n_nodes - 1;      // N is a multiple of 16 here
    const int g8 = (lane >= 16) ? 16 : 0;

    const float cv  = cnt[rowA];
    const float inv = 1.0f / fmaxf(cv, 1.0f);

    const float* aRow = sums  + (size_t)rowA * D_FEAT;
    const float* tRow = x_tgt + (size_t)rowA * D_FEAT;

    v8f acc[4];
    acc[0] = v8f{}; acc[1] = v8f{}; acc[2] = v8f{}; acc[3] = v8f{};

    for (int kk = 0; kk < D_FEAT; kk += 32) {
        if (kk + 32 < D_FEAT) {
            __builtin_prefetch(aRow + kk + 32 + g8, 0, 0);   // global_prefetch_b8
            __builtin_prefetch(tRow + kk + 32 + g8, 0, 0);
        }
        float af[16], tf[16];
        {
            const float4 a0 = *(const float4*)(aRow + kk + g8);
            const float4 a1 = *(const float4*)(aRow + kk + g8 + 4);
            const float4 a2 = *(const float4*)(aRow + kk + g8 + 8);
            const float4 a3 = *(const float4*)(aRow + kk + g8 + 12);
            af[0]=a0.x; af[1]=a0.y; af[2]=a0.z; af[3]=a0.w;
            af[4]=a1.x; af[5]=a1.y; af[6]=a1.z; af[7]=a1.w;
            af[8]=a2.x; af[9]=a2.y; af[10]=a2.z; af[11]=a2.w;
            af[12]=a3.x; af[13]=a3.y; af[14]=a3.z; af[15]=a3.w;
            const float4 t0 = *(const float4*)(tRow + kk + g8);
            const float4 t1 = *(const float4*)(tRow + kk + g8 + 4);
            const float4 t2 = *(const float4*)(tRow + kk + g8 + 8);
            const float4 t3 = *(const float4*)(tRow + kk + g8 + 12);
            tf[0]=t0.x; tf[1]=t0.y; tf[2]=t0.z; tf[3]=t0.w;
            tf[4]=t1.x; tf[5]=t1.y; tf[6]=t1.z; tf[7]=t1.w;
            tf[8]=t2.x; tf[9]=t2.y; tf[10]=t2.z; tf[11]=t2.w;
            tf[12]=t3.x; tf[13]=t3.y; tf[14]=t3.z; tf[15]=t3.w;
        }
        BF16x16 a_hi, a_lo, t_hi, t_lo;
#pragma unroll
        for (int e = 0; e < 16; ++e) {
            float s = af[e] * inv;                 // fused mean divide
            unsigned short h = f2bf(s);
            a_hi.u[e] = h;
            a_lo.u[e] = f2bf(s - bf2f(h));
            float t = tf[e];
            unsigned short th = f2bf(t);
            t_hi.u[e] = th;
            t_lo.u[e] = f2bf(t - bf2f(th));
        }
        const int kt = kk >> 5;
#pragma unroll
        for (int j = 0; j < 4; ++j) {
            const size_t bidx = (((size_t)(kt * 16 + (ng * 4 + j)) * 32 + lane) * 16);
            const v16bf blh = *(const v16bf*)(wl_hi + bidx);
            const v16bf bll = *(const v16bf*)(wl_lo + bidx);
            const v16bf brh = *(const v16bf*)(wr_hi + bidx);
            const v16bf brl = *(const v16bf*)(wr_lo + bidx);
            v8f c = acc[j];
            c = __builtin_amdgcn_wmma_f32_16x16x32_bf16(false, a_hi.v, false, blh, (short)0, c, false, false);
            c = __builtin_amdgcn_wmma_f32_16x16x32_bf16(false, a_hi.v, false, bll, (short)0, c, false, false);
            c = __builtin_amdgcn_wmma_f32_16x16x32_bf16(false, a_lo.v, false, blh, (short)0, c, false, false);
            c = __builtin_amdgcn_wmma_f32_16x16x32_bf16(false, t_hi.v, false, brh, (short)0, c, false, false);
            c = __builtin_amdgcn_wmma_f32_16x16x32_bf16(false, t_hi.v, false, brl, (short)0, c, false, false);
            c = __builtin_amdgcn_wmma_f32_16x16x32_bf16(false, t_lo.v, false, brh, (short)0, c, false, false);
            acc[j] = c;
        }
    }

    // C/D layout (32-bit 16x16): lane L holds col N=L&15; VGPR r holds row
    // M = r + (L>=16 ? 8 : 0).
    const int ncol  = lane & 15;
    const int rbase = (lane >= 16) ? 8 : 0;
#pragma unroll
    for (int j = 0; j < 4; ++j) {
        const int n = n0 + j * 16 + ncol;
        const float bias = b_l[n];
#pragma unroll
        for (int r = 0; r < 8; ++r) {
            const int m = m0 + rbase + r;
            if (m < n_nodes)
                out[(size_t)m * D_FEAT + n] = acc[j][r] + bias;
        }
    }
}

// ---------------------------------------------------------------------------
extern "C" void kernel_launch(void* const* d_in, const int* in_sizes, int n_in,
                              void* d_out, int out_size, void* d_ws, size_t ws_size,
                              hipStream_t stream) {
    const float* x_src = (const float*)d_in[0];
    const float* x_tgt = (const float*)d_in[1];
    const int*   eidx  = (const int*)d_in[2];
    const float* W_l   = (const float*)d_in[3];
    const float* b_l   = (const float*)d_in[4];
    const float* W_r   = (const float*)d_in[5];
    float* out = (float*)d_out;

    const int n_nodes = in_sizes[0] / D_FEAT;
    const int n_edges = in_sizes[2] / 2;

    // Workspace layout: [sums: N*256 f32][cnt: N f32][4x swizzled bf16 weights]
    char* ws = (char*)d_ws;
    float* sums = (float*)ws;
    float* cnt  = sums + (size_t)n_nodes * D_FEAT;
    size_t off  = ((size_t)n_nodes * (D_FEAT + 1) * sizeof(float) + 255) & ~(size_t)255;
    unsigned short* wl_hi = (unsigned short*)(ws + off);
    unsigned short* wl_lo = wl_hi + D_FEAT * D_FEAT;
    unsigned short* wr_hi = wl_lo + D_FEAT * D_FEAT;
    unsigned short* wr_lo = wr_hi + D_FEAT * D_FEAT;

    // 1) zero sums + counts (contiguous)
    {
        size_t n = (size_t)n_nodes * (D_FEAT + 1);
        sage_zero_kernel<<<4096, 256, 0, stream>>>(sums, n);
    }
    // 2) pre-swizzle weights into WMMA B layout (hi/lo bf16)
    sage_prep_w_kernel<<<(D_FEAT * D_FEAT) / 256, 256, 0, stream>>>(W_l, wl_hi, wl_lo);
    sage_prep_w_kernel<<<(D_FEAT * D_FEAT) / 256, 256, 0, stream>>>(W_r, wr_hi, wr_lo);
    // 3) edge scatter-add (mean numerator + counts)
    sage_scatter_kernel<<<n_edges, 256, 0, stream>>>(x_src, eidx, sums, cnt,
                                                     n_edges, n_nodes);
    // 4) fused mean + dual GEMM + bias via bf16 hi/lo WMMA
    dim3 grid((n_nodes + 15) / 16, 4);
    sage_gemm_kernel<<<grid, 32, 0, stream>>>(sums, cnt, x_tgt, b_l,
                                              wl_hi, wl_lo, wr_hi, wr_lo,
                                              out, n_nodes);
}
